// WaveRNN_72937134621065
// MI455X (gfx1250) — matrix operand order repaired and checked by reading, not compiled, not run on hardware
//
#include <hip/hip_runtime.h>

// ---------------------------------------------------------------------------
// WaveRNN on MI455X (gfx1250): 16 cooperating workgroups, bf16 WMMA 16x16x32,
// weights resident in L2 (bf16, 8.1 MB), grid barriers per phase.
// ---------------------------------------------------------------------------

typedef __attribute__((ext_vector_type(16))) __bf16 v16bf;
typedef __attribute__((ext_vector_type(8)))  __bf16 v8bf;
typedef __attribute__((ext_vector_type(8)))  float  v8f;

#define B_    4
#define T_    4800
#define H_    512
#define NCLS_ 512
#define NWG   16
#define NTH   256
#define NWAVES 8

union ABPack { v16bf v; v8bf h[2]; };

// One 16x16 output tile, K accumulated in steps of 32 (bf16 WMMA).
// A: [16][Kpad] bf16 (activations, rows are batch, K-contiguous)
// W: [Ncols][Kpad] bf16 (weights, row-major => each output column K-contiguous)
__device__ __forceinline__ v8f wmma_tile_bf16(const __bf16* __restrict__ W,
                                              const __bf16* __restrict__ A,
                                              int Kpad, int ncol0) {
  const int lane = threadIdx.x & 31;
  const int r    = lane & 15;
  const int hi   = lane >> 4;
  v8f acc = {};
  // A fragment: lanes 0-15 row=r hold K {k..k+15}; lanes 16-31 {k+16..k+31}.
  const __bf16* ap = A + (size_t)r * Kpad + (hi ? 16 : 0);
  // B fragment: lane col = ncol0+r; lanes 0-15 K {k..k+15}; lanes 16-31 {k+16..k+31}.
  const __bf16* bp = W + (size_t)(ncol0 + r) * (size_t)Kpad + (hi ? 16 : 0);
  for (int k = 0; k < Kpad; k += 32) {
    ABPack a;
    a.h[0] = *(const v8bf*)(ap + k);
    a.h[1] = *(const v8bf*)(ap + k + 8);
    v16bf b = *(const v16bf*)(bp + k);
    acc = __builtin_amdgcn_wmma_f32_16x16x32_bf16(false, a.v, false, b,
                                                  (short)0, acc, false, false);
  }
  return acc;
}

__device__ __forceinline__ float sigmoidf_(float v) {
  return 1.f / (1.f + __expf(-v));
}

// Monotonic-counter grid barrier (L2 atomics). 4800*6*16 << 2^32.
__device__ __forceinline__ void gsync(unsigned* bar, unsigned* epoch) {
  __syncthreads();
  if (threadIdx.x == 0) {
    __threadfence();
    unsigned tgt = *epoch + NWG;
    *epoch = tgt;
    atomicAdd(bar, 1u);
    while (__hip_atomic_load(bar, __ATOMIC_RELAXED, __HIP_MEMORY_SCOPE_AGENT) < tgt) {
      __builtin_amdgcn_s_sleep(1);
    }
    __threadfence();
  }
  __syncthreads();
}

struct Params {
  const float *mel, *aux;
  const float *b_I, *g1bih, *g1bhh, *g2bih, *g2bhh, *f1b, *f2b, *f3b;
  const __bf16 *wI, *w1i, *w1h, *w2i, *w2h, *wf1, *wf2, *wf3;
  __bf16 *X;    // [16][512]  x
  __bf16 *X2;   // [16][544]  x+h1n | a1
  __bf16 *X3;   // [16][544]  x2+h2n | a2
  __bf16 *Y1;   // [16][544]  relu(fc1) | a3
  __bf16 *Y2;   // [16][512]  relu(fc2)
  __bf16 *H1;   // [2][16][512] double-buffered hidden 1
  __bf16 *H2;   // [2][16][512] double-buffered hidden 2
  float  *pmax; // [NWG][4] partial argmax value
  int    *pidx; // [NWG][4] partial argmax index (global column)
  unsigned *bar;
  float *out_samples;  // [4][4800]
  float *out_logits;   // [4][4800][512]
};

__global__ void __launch_bounds__(NTH, 1) wavernn_persistent(Params p) {
  const int tid  = threadIdx.x;
  const int wg   = blockIdx.x;
  const int wave = tid >> 5;
  const int lane = tid & 31;
  const int c0   = wg * 32;  // this WG owns h-columns [c0, c0+32)

  __shared__ __bf16 s_in[16 * 128];     // padded input vector (A for W_I)
  __shared__ float  s_gi[3][32][4];     // gate pre-activations (input part)
  __shared__ float  s_gh[3][32][4];     // gate pre-activations (hidden part)
  __shared__ float  s_xa[32][4];        // x (P2) / x3 (after P3)
  __shared__ float  s_xb[32][4];        // x2
  __shared__ float  s_h [32][4];        // old hidden at own cols
  __shared__ float  s_lg[32][4];        // logits at own cols
  __shared__ float  s_sample[4];

  if (tid < 4) s_sample[tid] = 0.f;
  unsigned epoch = 0;

  for (int t = 0; t < T_; ++t) {
    // ------------------ P1: x = [sample, mel, a0] @ W_I.T + b_I -----------
    __syncthreads();   // s_sample ready / s_in reusable
    for (int i = tid; i < 16 * 128; i += NTH) {
      int m = i >> 7, c = i & 127;
      float v = 0.f;
      if (m < B_) {
        if (c == 0)        v = s_sample[m];
        else if (c < 81)   v = p.mel[(size_t)(m * T_ + t) * 80  + (c - 1)];
        else if (c < 113)  v = p.aux[(size_t)(m * T_ + t) * 128 + (c - 81)];
      }
      s_in[i] = (__bf16)v;
    }
    __syncthreads();
    for (int j = wave; j < 2; j += NWAVES) {
      int ncol = c0 + j * 16;
      v8f acc = wmma_tile_bf16(p.wI, s_in, 128, ncol);
      if (lane < 16) {
        int n = ncol + lane, lc = n - c0;
        float bb = p.b_I[n];
        float x0 = acc[0] + bb, x1 = acc[1] + bb, x2 = acc[2] + bb, x3 = acc[3] + bb;
        s_xa[lc][0] = x0; s_xa[lc][1] = x1; s_xa[lc][2] = x2; s_xa[lc][3] = x3;
        p.X[0 * H_ + n] = (__bf16)x0; p.X[1 * H_ + n] = (__bf16)x1;
        p.X[2 * H_ + n] = (__bf16)x2; p.X[3 * H_ + n] = (__bf16)x3;
      }
    }
    // aux slices for later phases (independent of GEMM results)
    if (wg == 1) for (int i = tid; i < 128; i += NTH) {
      int m = i >> 5, c = i & 31;
      p.X2[m * 544 + 512 + c] = (__bf16)p.aux[(size_t)(m * T_ + t) * 128 + 32 + c];
    }
    if (wg == 2) for (int i = tid; i < 128; i += NTH) {
      int m = i >> 5, c = i & 31;
      p.X3[m * 544 + 512 + c] = (__bf16)p.aux[(size_t)(m * T_ + t) * 128 + 64 + c];
    }
    if (wg == 3) for (int i = tid; i < 128; i += NTH) {
      int m = i >> 5, c = i & 31;
      p.Y1[m * 544 + 512 + c] = (__bf16)p.aux[(size_t)(m * T_ + t) * 128 + 96 + c];
    }
    gsync(p.bar, &epoch);

    // ------------------ P2: GRU1 ------------------------------------------
    {
      const __bf16* Hc = p.H1 + (size_t)(t & 1)       * (16 * H_);
      __bf16*       Hn = p.H1 + (size_t)((t & 1) ^ 1) * (16 * H_);
      for (int i = tid; i < 128; i += NTH) {
        int m = i & 3, c = i >> 2;
        s_h[c][m] = (float)Hc[m * H_ + c0 + c];
      }
      for (int j = wave; j < 6; j += NWAVES) {       // (gate, subtile)
        int g = j >> 1, st = j & 1;
        int ncol = g * H_ + c0 + st * 16;
        v8f ai = wmma_tile_bf16(p.w1i, p.X, H_, ncol);
        v8f ah = wmma_tile_bf16(p.w1h, Hc,  H_, ncol);
        if (lane < 16) {
          int n = ncol + lane, lc = st * 16 + lane;
          float bi = p.g1bih[n], bh = p.g1bhh[n];
          s_gi[g][lc][0] = ai[0] + bi; s_gi[g][lc][1] = ai[1] + bi;
          s_gi[g][lc][2] = ai[2] + bi; s_gi[g][lc][3] = ai[3] + bi;
          s_gh[g][lc][0] = ah[0] + bh; s_gh[g][lc][1] = ah[1] + bh;
          s_gh[g][lc][2] = ah[2] + bh; s_gh[g][lc][3] = ah[3] + bh;
        }
      }
      __syncthreads();
      for (int i = tid; i < 128; i += NTH) {
        int m = i & 3, c = i >> 2, col = c0 + c;
        float r = sigmoidf_(s_gi[0][c][m] + s_gh[0][c][m]);
        float z = sigmoidf_(s_gi[1][c][m] + s_gh[1][c][m]);
        float n = tanhf(s_gi[2][c][m] + r * s_gh[2][c][m]);
        float h = (1.f - z) * n + z * s_h[c][m];
        Hn[m * H_ + col] = (__bf16)h;
        float x2 = s_xa[c][m] + h;
        s_xb[c][m] = x2;
        p.X2[m * 544 + col] = (__bf16)x2;
      }
    }
    gsync(p.bar, &epoch);

    // ------------------ P3: GRU2 ------------------------------------------
    {
      const __bf16* Hc = p.H2 + (size_t)(t & 1)       * (16 * H_);
      __bf16*       Hn = p.H2 + (size_t)((t & 1) ^ 1) * (16 * H_);
      for (int i = tid; i < 128; i += NTH) {
        int m = i & 3, c = i >> 2;
        s_h[c][m] = (float)Hc[m * H_ + c0 + c];
      }
      for (int j = wave; j < 6; j += NWAVES) {
        int g = j >> 1, st = j & 1;
        int ncol = g * H_ + c0 + st * 16;
        v8f ai = wmma_tile_bf16(p.w2i, p.X2, 544, ncol);
        v8f ah = wmma_tile_bf16(p.w2h, Hc,   H_,  ncol);
        if (lane < 16) {
          int n = ncol + lane, lc = st * 16 + lane;
          float bi = p.g2bih[n], bh = p.g2bhh[n];
          s_gi[g][lc][0] = ai[0] + bi; s_gi[g][lc][1] = ai[1] + bi;
          s_gi[g][lc][2] = ai[2] + bi; s_gi[g][lc][3] = ai[3] + bi;
          s_gh[g][lc][0] = ah[0] + bh; s_gh[g][lc][1] = ah[1] + bh;
          s_gh[g][lc][2] = ah[2] + bh; s_gh[g][lc][3] = ah[3] + bh;
        }
      }
      __syncthreads();
      for (int i = tid; i < 128; i += NTH) {
        int m = i & 3, c = i >> 2, col = c0 + c;
        float r = sigmoidf_(s_gi[0][c][m] + s_gh[0][c][m]);
        float z = sigmoidf_(s_gi[1][c][m] + s_gh[1][c][m]);
        float n = tanhf(s_gi[2][c][m] + r * s_gh[2][c][m]);
        float h = (1.f - z) * n + z * s_h[c][m];
        Hn[m * H_ + col] = (__bf16)h;
        float x3 = s_xb[c][m] + h;
        s_xa[c][m] = x3;
        p.X3[m * 544 + col] = (__bf16)x3;
      }
    }
    gsync(p.bar, &epoch);

    // ------------------ P4: fc1 + relu ------------------------------------
    for (int j = wave; j < 2; j += NWAVES) {
      int ncol = c0 + j * 16;
      v8f acc = wmma_tile_bf16(p.wf1, p.X3, 544, ncol);
      if (lane < 16) {
        int n = ncol + lane;
        float bb = p.f1b[n];
        float y0 = fmaxf(acc[0] + bb, 0.f), y1 = fmaxf(acc[1] + bb, 0.f);
        float y2 = fmaxf(acc[2] + bb, 0.f), y3 = fmaxf(acc[3] + bb, 0.f);
        p.Y1[0 * 544 + n] = (__bf16)y0; p.Y1[1 * 544 + n] = (__bf16)y1;
        p.Y1[2 * 544 + n] = (__bf16)y2; p.Y1[3 * 544 + n] = (__bf16)y3;
      }
    }
    gsync(p.bar, &epoch);

    // ------------------ P5: fc2 + relu ------------------------------------
    for (int j = wave; j < 2; j += NWAVES) {
      int ncol = c0 + j * 16;
      v8f acc = wmma_tile_bf16(p.wf2, p.Y1, 544, ncol);
      if (lane < 16) {
        int n = ncol + lane;
        float bb = p.f2b[n];
        float y0 = fmaxf(acc[0] + bb, 0.f), y1 = fmaxf(acc[1] + bb, 0.f);
        float y2 = fmaxf(acc[2] + bb, 0.f), y3 = fmaxf(acc[3] + bb, 0.f);
        p.Y2[0 * H_ + n] = (__bf16)y0; p.Y2[1 * H_ + n] = (__bf16)y1;
        p.Y2[2 * H_ + n] = (__bf16)y2; p.Y2[3 * H_ + n] = (__bf16)y3;
      }
    }
    gsync(p.bar, &epoch);

    // ------------------ P6: fc3 logits + partial argmax -------------------
    for (int j = wave; j < 2; j += NWAVES) {
      int ncol = c0 + j * 16;
      v8f acc = wmma_tile_bf16(p.wf3, p.Y2, H_, ncol);
      if (lane < 16) {
        int n = ncol + lane, lc = n - c0;
        float bb = p.f3b[n];
        float l0 = acc[0] + bb, l1 = acc[1] + bb, l2 = acc[2] + bb, l3 = acc[3] + bb;
        p.out_logits[((size_t)(0 * T_ + t)) * NCLS_ + n] = l0;
        p.out_logits[((size_t)(1 * T_ + t)) * NCLS_ + n] = l1;
        p.out_logits[((size_t)(2 * T_ + t)) * NCLS_ + n] = l2;
        p.out_logits[((size_t)(3 * T_ + t)) * NCLS_ + n] = l3;
        s_lg[lc][0] = l0; s_lg[lc][1] = l1; s_lg[lc][2] = l2; s_lg[lc][3] = l3;
      }
    }
    __syncthreads();
    if (tid < 4) {
      float best = -3.4e38f; int bi = 0;
      for (int c = 0; c < 32; ++c) {               // first-max tie-break
        float v = s_lg[c][tid];
        if (v > best) { best = v; bi = c0 + c; }
      }
      p.pmax[wg * 4 + tid] = best;
      p.pidx[wg * 4 + tid] = bi;
    }
    gsync(p.bar, &epoch);

    // ------------------ P7: redundant global argmax -> sample -------------
    if (tid < 4) {
      float best = -3.4e38f; int bi = 0;
      for (int w = 0; w < NWG; ++w) {              // ascending => first max wins
        float v = p.pmax[w * 4 + tid];
        if (v > best) { best = v; bi = p.pidx[w * 4 + tid]; }
      }
      float samp = 2.f * (float)bi / (float)(NCLS_ - 1) - 1.f;
      s_sample[tid] = samp;
      if (wg == 0) p.out_samples[tid * T_ + t] = samp;
    }
    // top-of-loop __syncthreads orders s_sample for next P1
  }
}

// fp32 -> bf16 weight conversion with K padding to multiple of 32.
__global__ void cvt_pad_kernel(const float* __restrict__ src, __bf16* __restrict__ dst,
                               int N, int K, int Kpad) {
  int stride = gridDim.x * blockDim.x;
  int total = N * Kpad;
  for (int i = blockIdx.x * blockDim.x + threadIdx.x; i < total; i += stride) {
    int n = i / Kpad, k = i - n * Kpad;
    dst[i] = (k < K) ? (__bf16)src[(size_t)n * K + k] : (__bf16)0.f;
  }
}

extern "C" void kernel_launch(void* const* d_in, const int* in_sizes, int n_in,
                              void* d_out, int out_size, void* d_ws, size_t ws_size,
                              hipStream_t stream) {
  (void)in_sizes; (void)n_in; (void)out_size; (void)ws_size;
  const float* mel   = (const float*)d_in[0];
  const float* aux   = (const float*)d_in[1];
  const float* W_I   = (const float*)d_in[2];
  const float* b_I   = (const float*)d_in[3];
  const float* g1wih = (const float*)d_in[4];
  const float* g1whh = (const float*)d_in[5];
  const float* g1bih = (const float*)d_in[6];
  const float* g1bhh = (const float*)d_in[7];
  const float* g2wih = (const float*)d_in[8];
  const float* g2whh = (const float*)d_in[9];
  const float* g2bih = (const float*)d_in[10];
  const float* g2bhh = (const float*)d_in[11];
  const float* f1w   = (const float*)d_in[12];
  const float* f1b   = (const float*)d_in[13];
  const float* f2w   = (const float*)d_in[14];
  const float* f2b   = (const float*)d_in[15];
  const float* f3w   = (const float*)d_in[16];
  const float* f3b   = (const float*)d_in[17];

  char* ws = (char*)d_ws;
  size_t off = 0;
  auto alloc = [&](size_t bytes) {
    char* ptr = ws + off;
    off = (off + bytes + 255) & ~(size_t)255;
    return ptr;
  };
  __bf16* wI  = (__bf16*)alloc((size_t)512  * 128 * 2);
  __bf16* w1i = (__bf16*)alloc((size_t)1536 * 512 * 2);
  __bf16* w1h = (__bf16*)alloc((size_t)1536 * 512 * 2);
  __bf16* w2i = (__bf16*)alloc((size_t)1536 * 544 * 2);
  __bf16* w2h = (__bf16*)alloc((size_t)1536 * 512 * 2);
  __bf16* wf1 = (__bf16*)alloc((size_t)512  * 544 * 2);
  __bf16* wf2 = (__bf16*)alloc((size_t)512  * 544 * 2);
  __bf16* wf3 = (__bf16*)alloc((size_t)512  * 512 * 2);
  __bf16* X   = (__bf16*)alloc((size_t)16 * 512 * 2);
  __bf16* X2  = (__bf16*)alloc((size_t)16 * 544 * 2);
  __bf16* X3  = (__bf16*)alloc((size_t)16 * 544 * 2);
  __bf16* Y1  = (__bf16*)alloc((size_t)16 * 544 * 2);
  __bf16* Y2  = (__bf16*)alloc((size_t)16 * 512 * 2);
  __bf16* H1  = (__bf16*)alloc((size_t)2 * 16 * 512 * 2);
  __bf16* H2  = (__bf16*)alloc((size_t)2 * 16 * 512 * 2);
  float*  pmax = (float*)alloc((size_t)NWG * 4 * 4);
  int*    pidx = (int*)  alloc((size_t)NWG * 4 * 4);
  unsigned* bar = (unsigned*)alloc(256);
  size_t used = off;

  // Zero everything we use: barrier counter, initial hidden state, pad rows.
  hipMemsetAsync(d_ws, 0, used, stream);

  dim3 cg(128), cb(256);
  cvt_pad_kernel<<<cg, cb, 0, stream>>>(W_I,   wI,  512,  113, 128);
  cvt_pad_kernel<<<cg, cb, 0, stream>>>(g1wih, w1i, 1536, 512, 512);
  cvt_pad_kernel<<<cg, cb, 0, stream>>>(g1whh, w1h, 1536, 512, 512);
  cvt_pad_kernel<<<cg, cb, 0, stream>>>(g2wih, w2i, 1536, 544, 544);
  cvt_pad_kernel<<<cg, cb, 0, stream>>>(g2whh, w2h, 1536, 512, 512);
  cvt_pad_kernel<<<cg, cb, 0, stream>>>(f1w,   wf1, 512,  544, 544);
  cvt_pad_kernel<<<cg, cb, 0, stream>>>(f2w,   wf2, 512,  544, 544);
  cvt_pad_kernel<<<cg, cb, 0, stream>>>(f3w,   wf3, 512,  512, 512);

  Params p;
  p.mel = mel; p.aux = aux;
  p.b_I = b_I; p.g1bih = g1bih; p.g1bhh = g1bhh;
  p.g2bih = g2bih; p.g2bhh = g2bhh;
  p.f1b = f1b; p.f2b = f2b; p.f3b = f3b;
  p.wI = wI; p.w1i = w1i; p.w1h = w1h; p.w2i = w2i; p.w2h = w2h;
  p.wf1 = wf1; p.wf2 = wf2; p.wf3 = wf3;
  p.X = X; p.X2 = X2; p.X3 = X3; p.Y1 = Y1; p.Y2 = Y2;
  p.H1 = H1; p.H2 = H2;
  p.pmax = pmax; p.pidx = pidx; p.bar = bar;
  p.out_samples = (float*)d_out;
  p.out_logits  = (float*)d_out + (size_t)B_ * T_;

  wavernn_persistent<<<NWG, NTH, 0, stream>>>(p);
}
